// SparseCMLILoss_4140348473739
// MI455X (gfx1250) — matrix operand order repaired and checked by reading, not compiled, not run on hardware
//
#include <hip/hip_runtime.h>
#include <hip/hip_bf16.h>
#include <stdint.h>

typedef __attribute__((ext_vector_type(16))) __bf16 v16bf;
typedef __attribute__((ext_vector_type(8)))  __bf16 v8bf;
typedef __attribute__((ext_vector_type(8)))  float  v8f;

#define BATCH 256
#define SI 197
#define ST 64
#define DIM 768
#define NI 48      // image tokens used (ranks 1..48)
#define NT 15      // text tokens used (ranks 1..15), padded to 16
#define KSEL_I 49
#define KSEL_T 16

__device__ __forceinline__ unsigned short f32_to_bf16(float f) {
  unsigned u = __float_as_uint(f);
  u += 0x7FFFu + ((u >> 16) & 1u);           // round-to-nearest-even
  return (unsigned short)(u >> 16);
}

__device__ __forceinline__ unsigned orderable(float f) {
  unsigned u = __float_as_uint(f);
  return (u & 0x80000000u) ? ~u : (u | 0x80000000u);
}

// ---------------- Kernel 1: image top-49 + gather (ranks 1..48) ----------------
__global__ void topk_gather_image(const float* __restrict__ feat,
                                  const float* __restrict__ attn,
                                  unsigned short* __restrict__ Ig,
                                  float* __restrict__ out0) {
  const int b = blockIdx.x;
  const int tid = threadIdx.x;
  __shared__ float vals[SI];
  __shared__ unsigned long long best;
  __shared__ int sel[KSEL_I];

  if (b == 0 && tid == 0) out0[0] = 0.0f;   // zero loss accumulator (stream-ordered)

  if (tid < SI)
    vals[tid] = (tid == 0) ? -3.0e38f : attn[(size_t)b * SI * SI + tid];
  __syncthreads();

  for (int r = 0; r < KSEL_I; ++r) {
    if (tid == 0) best = 0ull;
    __syncthreads();
    if (tid < SI) {
      unsigned long long key = ((unsigned long long)orderable(vals[tid]) << 32)
                             | (unsigned long long)(0xFFFFFFFFu - (unsigned)tid);
      atomicMax(&best, key);
    }
    __syncthreads();
    if (tid == 0) {
      int idx = (int)(0xFFFFFFFFu - (unsigned)(best & 0xFFFFFFFFull));
      sel[r] = idx;
      vals[idx] = -3.0e38f;
    }
    __syncthreads();
  }

  for (int e = tid; e < NI * DIM; e += blockDim.x) {
    int r = e / DIM, c = e % DIM;
    float v = feat[((size_t)b * SI + sel[r + 1]) * DIM + c];
    Ig[((size_t)b * NI + r) * DIM + c] = f32_to_bf16(v);
  }
}

// ---------------- Kernel 2: text EOS-mask + top-16 + gather (ranks 1..15, pad row 15) ----
__global__ void topk_gather_text(const float* __restrict__ feat,
                                 const float* __restrict__ attn,
                                 const int* __restrict__ pmask,
                                 unsigned short* __restrict__ Tg) {
  const int b = blockIdx.x;
  const int tid = threadIdx.x;
  __shared__ float vals[ST];
  __shared__ unsigned long long best;
  __shared__ int sel[KSEL_T];

  if (tid == 0) {
    int cnt = 0, lz = 0;
    for (int j = 0; j < ST; ++j)
      if (pmask[b * ST + j] == 0) { cnt++; lz = j; }
    best = (unsigned long long)((cnt > 0) ? lz : 0);  // stash last_zero
  }
  __syncthreads();
  const int last_zero = (int)best;

  if (tid < ST) {
    float v = attn[(size_t)b * ST * ST + tid];
    if (tid == last_zero) v = 0.0f;   // EOS mask
    if (tid == 0) v = -3.0e38f;       // excluded by [1:] slice
    vals[tid] = v;
  }
  __syncthreads();

  for (int r = 0; r < KSEL_T; ++r) {
    if (tid == 0) best = 0ull;
    __syncthreads();
    if (tid < ST) {
      unsigned long long key = ((unsigned long long)orderable(vals[tid]) << 32)
                             | (unsigned long long)(0xFFFFFFFFu - (unsigned)tid);
      atomicMax(&best, key);
    }
    __syncthreads();
    if (tid == 0) {
      int idx = (int)(0xFFFFFFFFu - (unsigned)(best & 0xFFFFFFFFull));
      sel[r] = idx;
      vals[idx] = -3.0e38f;
    }
    __syncthreads();
  }

  for (int e = tid; e < 16 * DIM; e += blockDim.x) {
    int r = e / DIM, c = e % DIM;
    unsigned short v = 0;  // row 15 = zero pad (excluded from reductions later)
    if (r < NT)
      v = f32_to_bf16(feat[((size_t)b * ST + sel[r + 1]) * DIM + c]);
    Tg[((size_t)b * 16 + r) * DIM + c] = v;
  }
}

// ---------------- Kernel 3: CMLI pairwise sim via bf16 WMMA + max/mean reductions ----
__global__ void __launch_bounds__(256, 1) cmli_kernel(
    const unsigned short* __restrict__ Tg,
    const unsigned short* __restrict__ Ig,
    const float* __restrict__ logit_scale,
    float* __restrict__ out) {
  const int tid  = threadIdx.x;
  const int wave = tid >> 5;
  const int lane = tid & 31;
  const int l16  = lane >> 4;
  const int m    = lane & 15;
  const int x     = blockIdx.x >> 5;                 // text batch
  const int y     = ((blockIdx.x & 31) << 3) + wave; // image batch (one per wave)

  __shared__ __align__(32) unsigned short sT[16 * DIM];  // 24 KB: Tg[x] tile
  __shared__ float sSim[8 * 16 * NI];                    // 24 KB: per-wave 16x48 sim

  {
    const uint4* src = (const uint4*)(Tg + (size_t)x * 16 * DIM);
    uint4* dst = (uint4*)sT;
    for (int i = tid; i < (16 * DIM) / 8; i += 256) dst[i] = src[i];
  }
  __syncthreads();

  const float scale = logit_scale[0];
  const unsigned short* Iy = Ig + (size_t)y * NI * DIM;

  v8f acc0 = {}, acc1 = {}, acc2 = {};
  for (int k0 = 0; k0 < DIM; k0 += 32) {
    // A fragment (16x32 bf16): lanes 0-15 K={0..15}, lanes 16-31 K={16..31}
    union { v16bf v; v8bf h[2]; } a;
    const int ka = k0 + 16 * l16;
    a.h[0] = *(const v8bf*)(sT + m * DIM + ka);
    a.h[1] = *(const v8bf*)(sT + m * DIM + ka + 8);
    // B fragments (32x16 bf16): lane n=m holds 16 contiguous K at 16*l16
    const int kb = k0 + 16 * l16;
    v16bf b0 = *(const v16bf*)(Iy + ( 0 + m) * DIM + kb);
    v16bf b1 = *(const v16bf*)(Iy + (16 + m) * DIM + kb);
    v16bf b2 = *(const v16bf*)(Iy + (32 + m) * DIM + kb);
    acc0 = __builtin_amdgcn_wmma_f32_16x16x32_bf16(false, a.v, false, b0, (short)0, acc0, false, false);
    acc1 = __builtin_amdgcn_wmma_f32_16x16x32_bf16(false, a.v, false, b1, (short)0, acc1, false, false);
    acc2 = __builtin_amdgcn_wmma_f32_16x16x32_bf16(false, a.v, false, b2, (short)0, acc2, false, false);
  }

  // D layout: element e -> (row = e + 8*l16, col = tile*16 + m)
  float* simw = sSim + wave * (16 * NI);
  #pragma unroll
  for (int e = 0; e < 8; ++e) {
    const int row = e + 8 * l16;
    simw[row * NI + ( 0 + m)] = acc0[e] * scale;
    simw[row * NI + (16 + m)] = acc1[e] * scale;
    simw[row * NI + (32 + m)] = acc2[e] * scale;
  }
  __syncthreads();

  // i2t: mean over 48 image cols of (max over text rows 0..14)
  float csum = 0.0f;
  for (int col = lane; col < NI; col += 32) {
    float cm = -3.0e38f;
    for (int row = 0; row < NT; ++row) cm = fmaxf(cm, simw[row * NI + col]);
    csum += cm;
  }
  // t2i: mean over text rows 0..14 of (max over 48 image cols)
  float rsum = 0.0f;
  if (lane < NT) {
    float rm = -3.0e38f;
    for (int col = 0; col < NI; ++col) rm = fmaxf(rm, simw[lane * NI + col]);
    rsum = rm;
  }
  for (int o = 16; o > 0; o >>= 1) {
    csum += __shfl_xor(csum, o, 32);
    rsum += __shfl_xor(rsum, o, 32);
  }
  if (lane == 0) {
    out[1 + (size_t)x * BATCH + y]                 = csum * (1.0f / NI); // logits_per_image
    out[1 + BATCH * BATCH + (size_t)x * BATCH + y] = rsum * (1.0f / NT); // logits_per_text
  }
}

// ---------------- Kernel 4: CE on diagonals -> scalar loss ----------------
__global__ void ce_loss_kernel(const float* __restrict__ outbuf,
                               float* __restrict__ loss) {
  const int r   = blockIdx.x & (BATCH - 1);
  const int mtx = blockIdx.x >> 8;
  const float* row = outbuf + 1 + (size_t)mtx * BATCH * BATCH + (size_t)r * BATCH;
  const int tid = threadIdx.x;
  __shared__ float red[16];

  float v = row[tid];
  float mx = v;
  for (int o = 16; o > 0; o >>= 1) mx = fmaxf(mx, __shfl_xor(mx, o, 32));
  if ((tid & 31) == 0) red[tid >> 5] = mx;
  __syncthreads();
  mx = red[0];
  #pragma unroll
  for (int i = 1; i < 8; ++i) mx = fmaxf(mx, red[i]);

  float p = __expf(v - mx);
  for (int o = 16; o > 0; o >>= 1) p += __shfl_xor(p, o, 32);
  if ((tid & 31) == 0) red[8 + (tid >> 5)] = p;
  __syncthreads();
  if (tid == 0) {
    float s = 0.0f;
    #pragma unroll
    for (int i = 0; i < 8; ++i) s += red[8 + i];
    float logp = (row[r] - mx) - __logf(s);
    atomicAdd(loss, -logp * (1.0f / (2.0f * BATCH)));  // (ce_i + ce_t)/2, mean over rows
  }
}

extern "C" void kernel_launch(void* const* d_in, const int* in_sizes, int n_in,
                              void* d_out, int out_size, void* d_ws, size_t ws_size,
                              hipStream_t stream) {
  const float* image_features = (const float*)d_in[0];
  const float* text_features  = (const float*)d_in[1];
  const int*   padding_mask   = (const int*)d_in[2];
  const float* image_attn     = (const float*)d_in[3];
  const float* text_attn      = (const float*)d_in[4];
  const float* logit_scale    = (const float*)d_in[5];
  float* out = (float*)d_out;

  unsigned short* Tg = (unsigned short*)d_ws;                               // B*16*D bf16
  unsigned short* Ig = (unsigned short*)((char*)d_ws + (size_t)BATCH * 16 * DIM * 2);

  topk_gather_image<<<BATCH, 256, 0, stream>>>(image_features, image_attn, Ig, out);
  topk_gather_text <<<BATCH, 256, 0, stream>>>(text_features, text_attn, padding_mask, Tg);
  cmli_kernel      <<<BATCH * 32, 256, 0, stream>>>(Tg, Ig, logit_scale, out);
  ce_loss_kernel   <<<2 * BATCH, 256, 0, stream>>>(out, out);
}
